// MultiHeadAttention_48902497632402
// MI455X (gfx1250) — matrix operand order repaired and checked by reading, not compiled, not run on hardware
//
#include <hip/hip_runtime.h>
#include <hip/hip_bf16.h>

#define B_   8
#define S_   1024
#define DIM_ 768
#define H_   16
#define DH_  48
#define DHP_ 64

typedef __bf16 bf16x16 __attribute__((ext_vector_type(16)));
typedef float  f32x8   __attribute__((ext_vector_type(8)));

union AFragU { bf16x16 v; unsigned short u[16]; };

__device__ __forceinline__ unsigned short f2bf(float f) {
  unsigned u = __float_as_uint(f);
  u += 0x7FFFu + ((u >> 16) & 1u);   // round-to-nearest-even
  return (unsigned short)(u >> 16);
}

// A fragment (16x32 bf16): lane&15 = row M,
// lane half selects contiguous K chunk (0-15 vs 16-31), packed in pairs.
__device__ __forceinline__ bf16x16 load_fragA(const unsigned short* p, int ld) {
  int lane = threadIdx.x & 31;
  int r  = lane & 15;
  int kh = (lane >> 4) & 1;
  AFragU a;
#pragma unroll
  for (int i = 0; i < 8; ++i) {
    int kb = (kh << 4) + (i << 1);
    const unsigned short* q = p + r * ld + kb;
    a.u[2 * i]     = q[0];
    a.u[2 * i + 1] = q[1];
  }
  return a.v;
}

// B fragment (32x16 bf16) from N-major [n][k] storage, same K order as load_fragA:
// lane&15 = column N, lane half selects contiguous K chunk (0-15 vs 16-31).
__device__ __forceinline__ bf16x16 load_fragB(const unsigned short* p, int ld) {
  int lane = threadIdx.x & 31;
  int n  = lane & 15;
  int kh = (lane >> 4) & 1;
  const unsigned short* q = p + n * ld + kh * 16;
  AFragU b;
#pragma unroll
  for (int j = 0; j < 16; ++j) b.u[j] = q[j];
  return b.v;
}

__device__ __forceinline__ f32x8 wmma_bf16(bf16x16 a, bf16x16 b, f32x8 c) {
  return __builtin_amdgcn_wmma_f32_16x16x32_bf16(false, a, false, b, (short)0, c, false, false);
}

// ---------------- conversion kernels ----------------

__global__ void k_cvt(const float* __restrict__ src, unsigned short* __restrict__ dst, int n) {
  int i = blockIdx.x * blockDim.x + threadIdx.x;
  if (i < n) dst[i] = f2bf(src[i]);
}

// Wq/Wk/Wv [H,768,48] f32 -> transposed bf16 [H,48,768] (N-major for B frags)
__global__ void k_cvt_wT(const float* __restrict__ src, unsigned short* __restrict__ dst) {
  int i = blockIdx.x * blockDim.x + threadIdx.x;
  int n = H_ * DH_ * DIM_;
  if (i >= n) return;
  int d = i % DIM_;
  int r = (i / DIM_) % DH_;
  int h = i / (DH_ * DIM_);
  dst[i] = f2bf(src[((size_t)h * DIM_ + d) * DH_ + r]);
}

// Wp [768,768] f32 -> transposed bf16 [768(n),768(k)]
__global__ void k_cvt_wpT(const float* __restrict__ src, unsigned short* __restrict__ dst) {
  int i = blockIdx.x * blockDim.x + threadIdx.x;
  if (i >= DIM_ * DIM_) return;
  int k = i % DIM_;
  int n = i / DIM_;
  dst[i] = f2bf(src[(size_t)k * DIM_ + n]);
}

// ---------------- QKV projection ----------------
// grid (S/16, H, B), block 96 (3 waves: Q,K,V)
// Q,K stored row-major [b,h,s,64] zero-padded; V stored transposed [b,h,64,S]

__global__ void __launch_bounds__(96) k_qkv(
    const unsigned short* __restrict__ xb,
    const unsigned short* __restrict__ wqT, const unsigned short* __restrict__ wkT,
    const unsigned short* __restrict__ wvT,
    const float* __restrict__ bq, const float* __restrict__ bk, const float* __restrict__ bv,
    unsigned short* __restrict__ Qb, unsigned short* __restrict__ Kb,
    unsigned short* __restrict__ Vt) {
  int wid  = threadIdx.x >> 5;
  int lane = threadIdx.x & 31;
  int s0 = blockIdx.x * 16;
  int h  = blockIdx.y;
  int b  = blockIdx.z;

  const unsigned short* WT   = (wid == 0) ? wqT : (wid == 1) ? wkT : wvT;
  const float*          bias = (wid == 0) ? bq : (wid == 1) ? bk : bv;

  const unsigned short* xrow  = xb + ((size_t)b * S_ + s0) * DIM_;
  const unsigned short* wbase = WT + (size_t)h * DH_ * DIM_;

  f32x8 c0 = {}, c1 = {}, c2 = {};
  for (int k0 = 0; k0 < DIM_; k0 += 32) {
    bf16x16 a  = load_fragA(xrow + k0, DIM_);
    bf16x16 b0 = load_fragB(wbase + (size_t)0  * DIM_ + k0, DIM_);
    bf16x16 b1 = load_fragB(wbase + (size_t)16 * DIM_ + k0, DIM_);
    bf16x16 b2 = load_fragB(wbase + (size_t)32 * DIM_ + k0, DIM_);
    c0 = wmma_bf16(a, b0, c0);
    c1 = wmma_bf16(a, b1, c1);
    c2 = wmma_bf16(a, b2, c2);
  }
  int n  = lane & 15;
  int mh = (lane >> 4) & 1;
  float bb0 = bias[h * DH_ + n];
  float bb1 = bias[h * DH_ + 16 + n];
  float bb2 = bias[h * DH_ + 32 + n];
  size_t bh = (size_t)b * H_ + h;

  if (wid == 2) {
    // V transposed: [bh, e, t]
    unsigned short* vrow = Vt + bh * DHP_ * S_;
#pragma unroll
    for (int i = 0; i < 8; ++i) {
      int t = s0 + mh * 8 + i;
      vrow[(size_t)(n)      * S_ + t] = f2bf(c0[i] + bb0);
      vrow[(size_t)(16 + n) * S_ + t] = f2bf(c1[i] + bb1);
      vrow[(size_t)(32 + n) * S_ + t] = f2bf(c2[i] + bb2);
    }
  } else {
    unsigned short* drow = ((wid == 0) ? Qb : Kb) + (bh * S_ + s0) * DHP_;
#pragma unroll
    for (int i = 0; i < 8; ++i) {
      int m = mh * 8 + i;
      drow[m * DHP_ + n]      = f2bf(c0[i] + bb0);
      drow[m * DHP_ + 16 + n] = f2bf(c1[i] + bb1);
      drow[m * DHP_ + 32 + n] = f2bf(c2[i] + bb2);
      drow[m * DHP_ + 48 + n] = 0;   // pad cols are K-dim of score WMMAs: must be 0
    }
  }
}

// ---------------- flash attention (transposed-score formulation) ----------------
// grid (S/16, H, B), block 32.  ST = K.Q^T so s lives in the lane dim and t in
// the register dim: softmax is in-register + one shfl_xor(16); P^T feeds the
// next WMMA as a B operand via 16 half-swap shuffles (no LDS, no barriers).

__global__ void __launch_bounds__(32) k_attn(
    const unsigned short* __restrict__ Qb, const unsigned short* __restrict__ Kb,
    const unsigned short* __restrict__ Vt, unsigned short* __restrict__ attn) {
  int lane = threadIdx.x & 31;
  int s0 = blockIdx.x * 16;
  int h  = blockIdx.y;
  int b  = blockIdx.z;
  size_t bh = (size_t)b * H_ + h;
  const unsigned short* qbase = Qb + (bh * S_ + s0) * DHP_;
  const unsigned short* kbase = Kb + bh * S_ * DHP_;
  const unsigned short* vbase = Vt + bh * DHP_ * S_;   // [e][t]

  // Q as B operand: lane&15 = s, contiguous e per lane
  bf16x16 qb0 = load_fragB(qbase + 0,  DHP_);
  bf16x16 qb1 = load_fragB(qbase + 32, DHP_);

  f32x8 o0 = {}, o1 = {}, o2 = {};   // O^T tiles: reg dim = e, lane dim = s
  float mrun = -3.0e38f, lrun = 0.0f; // per-lane stats for s = lane&15

  const float rscale = 0.03608439182435161f;  // 1/sqrt(768)
  int n  = lane & 15;
  int mh = (lane >> 4) & 1;

  for (int t0 = 0; t0 < S_; t0 += 32) {
    if (t0 + 32 < S_) {
      __builtin_prefetch(kbase + (size_t)(t0 + 32) * DHP_, 0, 1);
      __builtin_prefetch(vbase + t0 + 32, 0, 1);
    }
    // ST = K.Q^T : two 16x16 tiles (t rows t0..t0+15, t0+16..t0+31)
    f32x8 st0 = {}, st1 = {};
    bf16x16 ka0a = load_fragA(kbase + (size_t)(t0 + 0)  * DHP_ + 0,  DHP_);
    bf16x16 ka0b = load_fragA(kbase + (size_t)(t0 + 0)  * DHP_ + 32, DHP_);
    bf16x16 ka1a = load_fragA(kbase + (size_t)(t0 + 16) * DHP_ + 0,  DHP_);
    bf16x16 ka1b = load_fragA(kbase + (size_t)(t0 + 16) * DHP_ + 32, DHP_);
    st0 = wmma_bf16(ka0a, qb0, st0);
    st0 = wmma_bf16(ka0b, qb1, st0);
    st1 = wmma_bf16(ka1a, qb0, st1);
    st1 = wmma_bf16(ka1b, qb1, st1);

    // ---- online softmax over t (register dim) ----
#pragma unroll
    for (int i = 0; i < 8; ++i) { st0[i] *= rscale; st1[i] *= rscale; }
    float mx = fmaxf(st0[0], st1[0]);
#pragma unroll
    for (int i = 1; i < 8; ++i) mx = fmaxf(mx, fmaxf(st0[i], st1[i]));
    mx = fmaxf(mx, __shfl_xor(mx, 16, 32));   // merge the two t-halves
    float mnew  = fmaxf(mrun, mx);
    float alpha = __expf(mrun - mnew);
    float rs = 0.0f;
#pragma unroll
    for (int i = 0; i < 8; ++i) {
      st0[i] = __expf(st0[i] - mnew); rs += st0[i];
      st1[i] = __expf(st1[i] - mnew); rs += st1[i];
    }
    rs += __shfl_xor(rs, 16, 32);
    lrun = lrun * alpha + rs;
    mrun = mnew;
#pragma unroll
    for (int i = 0; i < 8; ++i) { o0[i] *= alpha; o1[i] *= alpha; o2[i] *= alpha; }

    // ---- pack P^T (C layout) into B-frag layout via half swaps ----
    // C layout: lane(mh) holds t = 8*mh+i (st0) and 16+8*mh+i (st1)
    // B layout: lane(kh) needs t = 16*kh + j, j = 0..15
    AFragU pb;
#pragma unroll
    for (int i = 0; i < 8; ++i) {
      float x0 = __shfl_xor(st0[i], 16, 32);  // other half's st0[i]
      float x1 = __shfl_xor(st1[i], 16, 32);  // other half's st1[i]
      float lo = mh ? x1      : st0[i];       // j = i
      float hi = mh ? st1[i]  : x0;           // j = 8 + i
      pb.u[i]     = f2bf(lo);
      pb.u[8 + i] = f2bf(hi);
    }

    // ---- O^T += V^T . P^T ----
    bf16x16 va0 = load_fragA(vbase + (size_t)0  * S_ + t0, S_);
    bf16x16 va1 = load_fragA(vbase + (size_t)16 * S_ + t0, S_);
    bf16x16 va2 = load_fragA(vbase + (size_t)32 * S_ + t0, S_);
    o0 = wmma_bf16(va0, pb.v, o0);
    o1 = wmma_bf16(va1, pb.v, o1);
    o2 = wmma_bf16(va2, pb.v, o2);
  }

  // normalize (scalar per lane) and store contiguous 16B runs:
  // lane holds O^T[e = 16j + 8*mh + i][s = n]
  float inv = 1.0f / lrun;
  unsigned short* arow = attn + ((size_t)b * S_ + s0 + n) * DIM_ + h * DH_ + mh * 8;
  union { unsigned short us[8]; uint4 q; } pk;
#pragma unroll
  for (int i = 0; i < 8; ++i) pk.us[i] = f2bf(o0[i] * inv);
  *(uint4*)(arow + 0)  = pk.q;
#pragma unroll
  for (int i = 0; i < 8; ++i) pk.us[i] = f2bf(o1[i] * inv);
  *(uint4*)(arow + 16) = pk.q;
#pragma unroll
  for (int i = 0; i < 8; ++i) pk.us[i] = f2bf(o2[i] * inv);
  *(uint4*)(arow + 32) = pk.q;
}

// ---------------- output projection ----------------
// grid (DIM/64, B*S/16), block 32; each wave -> 16x64 tile of d_out

__global__ void __launch_bounds__(32) k_proj(
    const unsigned short* __restrict__ attn, const unsigned short* __restrict__ wpT,
    const float* __restrict__ bp, float* __restrict__ out) {
  int lane = threadIdx.x & 31;
  int n0 = blockIdx.x * 64;
  int m0 = blockIdx.y * 16;

  const unsigned short* arow = attn + (size_t)m0 * DIM_;
  f32x8 c0 = {}, c1 = {}, c2 = {}, c3 = {};
  for (int k0 = 0; k0 < DIM_; k0 += 32) {
    bf16x16 a  = load_fragA(arow + k0, DIM_);
    bf16x16 b0 = load_fragB(wpT + (size_t)(n0 + 0)  * DIM_ + k0, DIM_);
    bf16x16 b1 = load_fragB(wpT + (size_t)(n0 + 16) * DIM_ + k0, DIM_);
    bf16x16 b2 = load_fragB(wpT + (size_t)(n0 + 32) * DIM_ + k0, DIM_);
    bf16x16 b3 = load_fragB(wpT + (size_t)(n0 + 48) * DIM_ + k0, DIM_);
    c0 = wmma_bf16(a, b0, c0);
    c1 = wmma_bf16(a, b1, c1);
    c2 = wmma_bf16(a, b2, c2);
    c3 = wmma_bf16(a, b3, c3);
  }
  int n  = lane & 15;
  int mh = (lane >> 4) & 1;
  float bb0 = bp[n0 + n];
  float bb1 = bp[n0 + 16 + n];
  float bb2 = bp[n0 + 32 + n];
  float bb3 = bp[n0 + 48 + n];
  float* orow = out + (size_t)m0 * DIM_ + n0;
#pragma unroll
  for (int i = 0; i < 8; ++i) {
    int m = mh * 8 + i;
    orow[m * DIM_ + n]      = c0[i] + bb0;
    orow[m * DIM_ + 16 + n] = c1[i] + bb1;
    orow[m * DIM_ + 32 + n] = c2[i] + bb2;
    orow[m * DIM_ + 48 + n] = c3[i] + bb3;
  }
}

// ---------------- host launch ----------------

extern "C" void kernel_launch(void* const* d_in, const int* in_sizes, int n_in,
                              void* d_out, int out_size, void* d_ws, size_t ws_size,
                              hipStream_t stream) {
  const float* x  = (const float*)d_in[0];
  const float* Wq = (const float*)d_in[1];
  const float* bq = (const float*)d_in[2];
  const float* Wk = (const float*)d_in[3];
  const float* bk = (const float*)d_in[4];
  const float* Wv = (const float*)d_in[5];
  const float* bv = (const float*)d_in[6];
  const float* Wp = (const float*)d_in[7];
  const float* bp = (const float*)d_in[8];
  float* out = (float*)d_out;

  char* ws = (char*)d_ws;
  size_t off = 0;
  auto alloc = [&](size_t bytes) -> char* {
    char* p = ws + off;
    off += (bytes + 255) & ~(size_t)255;
    return p;
  };
  unsigned short* xb    = (unsigned short*)alloc((size_t)B_ * S_ * DIM_ * 2);
  unsigned short* wqT   = (unsigned short*)alloc((size_t)H_ * DH_ * DIM_ * 2);
  unsigned short* wkT   = (unsigned short*)alloc((size_t)H_ * DH_ * DIM_ * 2);
  unsigned short* wvT   = (unsigned short*)alloc((size_t)H_ * DH_ * DIM_ * 2);
  unsigned short* wpT   = (unsigned short*)alloc((size_t)DIM_ * DIM_ * 2);
  unsigned short* Qb    = (unsigned short*)alloc((size_t)B_ * H_ * S_ * DHP_ * 2);
  unsigned short* Kb    = (unsigned short*)alloc((size_t)B_ * H_ * S_ * DHP_ * 2);
  unsigned short* Vt    = (unsigned short*)alloc((size_t)B_ * H_ * DHP_ * S_ * 2);
  unsigned short* attnb = (unsigned short*)alloc((size_t)B_ * S_ * DIM_ * 2);

  int nx = B_ * S_ * DIM_;
  k_cvt<<<(nx + 255) / 256, 256, 0, stream>>>(x, xb, nx);
  int nwp = DIM_ * DIM_;
  k_cvt_wpT<<<(nwp + 255) / 256, 256, 0, stream>>>(Wp, wpT);
  int nw = H_ * DH_ * DIM_;
  k_cvt_wT<<<(nw + 255) / 256, 256, 0, stream>>>(Wq, wqT);
  k_cvt_wT<<<(nw + 255) / 256, 256, 0, stream>>>(Wk, wkT);
  k_cvt_wT<<<(nw + 255) / 256, 256, 0, stream>>>(Wv, wvT);

  dim3 gq(S_ / 16, H_, B_);
  k_qkv<<<gq, 96, 0, stream>>>(xb, wqT, wkT, wvT, bq, bk, bv, Qb, Kb, Vt);

  dim3 ga(S_ / 16, H_, B_);
  k_attn<<<ga, 32, 0, stream>>>(Qb, Kb, Vt, attnb);

  dim3 gp(DIM_ / 64, (B_ * S_) / 16, 1);
  k_proj<<<gp, 32, 0, stream>>>(attnb, wpT, bp, out);
}
